// EuclideanCodebook_89747636617343
// MI455X (gfx1250) — compile-verified, matrix operand order checked by reading, not run on hardware
//
#include <hip/hip_runtime.h>

// ---------------------------------------------------------------------------
// Euclidean codebook VQ for MI455X (gfx1250, wave32, WMMA + async-to-LDS).
//   argmax_k ( 2*x_n.e_k - ||e_k||^2 )   (||x||^2 is row-constant, drops out)
// GEMM via v_wmma_f32_16x16x32_bf16 (f32 accumulate).
// Codebook double-buffered through LDS with global_load_async_to_lds_b128.
// Each wave holds a 32-row A tile in VGPRs so every LDS B-load feeds 2 WMMAs.
// ---------------------------------------------------------------------------

typedef __attribute__((ext_vector_type(16))) __bf16 v16bf;
typedef __attribute__((ext_vector_type(8)))  float  v8f;

#define DIMS            128
#define CHUNK           128        // codes per LDS buffer
#define LDS_STRIDE      136        // bf16 per LDS row: 128 + 8 pad (bank dodge)
#define BUF_BYTES       (CHUNK * LDS_STRIDE * 2)   // 34816 B per buffer
#define ROWS_PER_WAVE   32
#define WAVES_PER_BLOCK 8
#define ROWS_PER_BLOCK  (ROWS_PER_WAVE * WAVES_PER_BLOCK)   // 256

__device__ __forceinline__ unsigned short f2bf(float f) {
  unsigned int u = __builtin_bit_cast(unsigned int, f);
  u += 0x7fffu + ((u >> 16) & 1u);          // round-to-nearest-even
  return (unsigned short)(u >> 16);
}

union VB {
  v16bf          v;
  unsigned short s[16];
  uint4          q[2];
};

// Async fill of one CHUNK buffer: 2048 x 16B, 8 b128 issues per thread.
// Each lane supplies its own (LDS offset, global addr); tracked by ASYNCcnt.
__device__ __forceinline__ void stage_chunk_async(const unsigned short* ebf_cb,
                                                  unsigned lds_base_bytes,
                                                  int tid) {
#pragma unroll
  for (int i = 0; i < 8; ++i) {
    const int cid  = tid + i * 256;         // 16B chunk id within CHUNK
    const int code = cid >> 4;              // 16 chunks per code row
    const int part = cid & 15;
    const unsigned lds_off =
        lds_base_bytes + (unsigned)(code * LDS_STRIDE + part * 8) * 2u;
    const void* g = (const void*)((const uint4*)ebf_cb + cid);
    asm volatile("global_load_async_to_lds_b128 %0, %1, off"
                 :: "v"(lds_off), "v"(g) : "memory");
  }
}

__device__ __forceinline__ void wait_async0() {
  asm volatile("s_wait_asynccnt 0x0" ::: "memory");
}

// ---------------------------------------------------------------------------
// Prep: embed f32 -> bf16 + per-code squared norms (tiny, L2-resident after).
// ---------------------------------------------------------------------------
__global__ void __launch_bounds__(256)
vq_prep_kernel(const float* __restrict__ embed,
               unsigned short* __restrict__ ebf,
               float* __restrict__ e2, int K) {
  int code = blockIdx.x * blockDim.x + threadIdx.x;
  if (code >= K) return;
  const float* row = embed + (size_t)code * DIMS;
  unsigned short* orow = ebf + (size_t)code * DIMS;
  float s = 0.f;
#pragma unroll 8
  for (int d = 0; d < DIMS; ++d) {
    float v = row[d];
    s += v * v;
    orow[d] = f2bf(v);
  }
  e2[code] = s;
}

// ---------------------------------------------------------------------------
__global__ void __launch_bounds__(256)
vq_main_kernel(const float* __restrict__ x,
               const float* __restrict__ embed,          // f32 (dequant copy)
               const unsigned short* __restrict__ ebf,   // bf16 codebook
               const float* __restrict__ e2g,            // ||e||^2
               int*   __restrict__ out_idx,
               float* __restrict__ out_q,
               int K) {
  extern __shared__ __align__(16) unsigned short smem[];  // 2 * BUF_BYTES dyn LDS
  const unsigned smem_base = (unsigned)(size_t)&smem[0];  // flat low32 == LDS off

  const int tid  = threadIdx.x;
  const int lane = tid & 31;
  const int wave = tid >> 5;
  const int rowbase = blockIdx.x * ROWS_PER_BLOCK + wave * ROWS_PER_WAVE;

  const int col = lane & 15;   // M for A / N for B-C-D
  const int h   = lane >> 4;   // K half selector

  // ---- Two A tiles (rows rowbase..+15 and +16..+31), ISA 16-bit A layout ---
  VB a[2][4];
#pragma unroll
  for (int m = 0; m < 2; ++m) {
    const float* xr = x + (size_t)(rowbase + m * 16 + col) * DIMS;
#pragma unroll
    for (int s = 0; s < 4; ++s) {
      const int k0 = s * 32 + h * 8;
#pragma unroll
      for (int j = 0; j < 8; ++j) a[m][s].s[j]     = f2bf(xr[k0 + j]);
#pragma unroll
      for (int j = 0; j < 8; ++j) a[m][s].s[8 + j] = f2bf(xr[k0 + 16 + j]);
    }
  }

  float bestv[2][8];
  int   besti[2][8];
#pragma unroll
  for (int m = 0; m < 2; ++m)
#pragma unroll
    for (int r = 0; r < 8; ++r) { bestv[m][r] = -3.402823466e+38f; besti[m][r] = 0; }

  // ---- Prologue: async-stage chunk 0 into buffer 0 -------------------------
  stage_chunk_async(ebf, smem_base, tid);
  wait_async0();
  __syncthreads();

  const int nchunks = K / CHUNK;
  for (int cc = 0; cc < nchunks; ++cc) {
    const int buf = cc & 1;
    // Kick off async fill of the next chunk into the other buffer, then
    // compute on the current one — staging hides behind WMMA work.
    if (cc + 1 < nchunks)
      stage_chunk_async(ebf + (size_t)(cc + 1) * CHUNK * DIMS,
                        smem_base + (1 - buf) * BUF_BYTES, tid);

    const unsigned short* bbase = smem + (size_t)buf * (BUF_BYTES / 2);
    const int cb = cc * CHUNK;

#pragma unroll 2
    for (int t = 0; t < CHUNK / 16; ++t) {
      const unsigned short* brow = bbase + (t * 16 + col) * LDS_STRIDE;
      v8f c0 = {}, c1 = {};
#pragma unroll
      for (int s = 0; s < 4; ++s) {
        VB b;
        const int k0 = s * 32 + h * 8;
        b.q[0] = *(const uint4*)&brow[k0];
        b.q[1] = *(const uint4*)&brow[k0 + 16];
        c0 = __builtin_amdgcn_wmma_f32_16x16x32_bf16(
                 false, a[0][s].v, false, b.v, (short)0, c0, false, false);
        c1 = __builtin_amdgcn_wmma_f32_16x16x32_bf16(
                 false, a[1][s].v, false, b.v, (short)0, c1, false, false);
      }
      const int   code_idx = cb + t * 16 + col;
      const float e2v = e2g[code_idx];
#pragma unroll
      for (int r = 0; r < 8; ++r) {
        const float v0 = 2.0f * c0[r] - e2v;
        if (v0 > bestv[0][r]) { bestv[0][r] = v0; besti[0][r] = code_idx; }
        const float v1 = 2.0f * c1[r] - e2v;
        if (v1 > bestv[1][r]) { bestv[1][r] = v1; besti[1][r] = code_idx; }
      }
    }

    wait_async0();        // next buffer fully in LDS (per-wave ASYNCcnt)
    __syncthreads();      // all waves done reading current buffer
  }

  // ---- Argmax reduction across the 16 column-lanes of each half ------------
#pragma unroll
  for (int m = 0; m < 2; ++m)
#pragma unroll
    for (int r = 0; r < 8; ++r)
#pragma unroll
      for (int mask = 1; mask < 16; mask <<= 1) {
        const float ov = __shfl_xor(bestv[m][r], mask, 32);
        const int   oi = __shfl_xor(besti[m][r], mask, 32);
        if (ov > bestv[m][r] || (ov == bestv[m][r] && oi < besti[m][r])) {
          bestv[m][r] = ov; besti[m][r] = oi;
        }
      }

  // ---- Write indices (int32, jnp.argmax tie order) -------------------------
  if (lane == 0) {
#pragma unroll
    for (int m = 0; m < 2; ++m)
#pragma unroll
      for (int r = 0; r < 8; ++r)
        out_idx[rowbase + m * 16 + r] = besti[m][r];
  } else if (lane == 16) {
#pragma unroll
    for (int m = 0; m < 2; ++m)
#pragma unroll
      for (int r = 0; r < 8; ++r)
        out_idx[rowbase + m * 16 + 8 + r] = besti[m][r];
  }

  // ---- Dequantize: cooperative float4 copy of winning f32 embed rows -------
#pragma unroll
  for (int rr = 0; rr < ROWS_PER_WAVE; ++rr) {
    const int m    = rr >> 4;
    const int slot = rr & 7;
    const int srcl = ((rr & 15) < 8) ? 0 : 16;
    const int bi   = __shfl(besti[m][slot], srcl, 32);
    const float4* src = (const float4*)(embed + (size_t)bi * DIMS);
    float4* dst = (float4*)(out_q + (size_t)(rowbase + rr) * DIMS);
    dst[lane] = src[lane];   // 32 lanes x 16B = 128 floats
  }
}

// ---------------------------------------------------------------------------
extern "C" void kernel_launch(void* const* d_in, const int* in_sizes, int n_in,
                              void* d_out, int out_size, void* d_ws, size_t ws_size,
                              hipStream_t stream) {
  const float* x     = (const float*)d_in[0];   // (16, 8192, 128) f32
  const float* embed = (const float*)d_in[1];   // (1024, 128) f32

  const int Nrows = in_sizes[0] / DIMS;         // 131072
  const int K     = in_sizes[1] / DIMS;         // 1024

  // Workspace: [ bf16 codebook (K*128*2 B) | e2 (K*4 B) ]
  unsigned short* ebf = (unsigned short*)d_ws;
  float*          e2  = (float*)((char*)d_ws + (size_t)K * DIMS * sizeof(unsigned short));

  // Output: [ int32 indices (Nrows) | f32 quantize (Nrows*128) ] flat
  int*   out_idx = (int*)d_out;
  float* out_q   = (float*)d_out + Nrows;

  vq_prep_kernel<<<(K + 255) / 256, 256, 0, stream>>>(embed, ebf, e2, K);
  vq_main_kernel<<<Nrows / ROWS_PER_BLOCK, 256, 2 * BUF_BYTES, stream>>>(
      x, embed, ebf, e2, out_idx, out_q, K);
}
